// GlobalPointer_29068338659862
// MI455X (gfx1250) — matrix operand order repaired and checked by reading, not compiled, not run on hardware
//
#include <hip/hip_runtime.h>
#include <hip/hip_bf16.h>

#define HEADS      12
#define HEAD_SIZE  64
#define B_DIM      8
#define L_DIM      1024
#define K_DIM      768          // hidden
#define N_DIM      1536         // HEADS * HEAD_SIZE * 2
#define M_DIM      (B_DIM * L_DIM)
#define MASK_VALF  1000000000000.0f

typedef _Float16 v8h  __attribute__((ext_vector_type(8)));
typedef _Float16 v16h __attribute__((ext_vector_type(16)));
typedef _Float16 h2   __attribute__((ext_vector_type(2)));
typedef float    v8f  __attribute__((ext_vector_type(8)));
typedef float    f4   __attribute__((ext_vector_type(4)));

#if defined(__has_builtin)
#if __has_builtin(__builtin_amdgcn_global_load_async_to_lds_b128)
#define HAVE_ASYNC_LDS 1
#else
#define HAVE_ASYNC_LDS 0
#endif
#if __has_builtin(__builtin_amdgcn_s_wait_asynccnt)
#define HAVE_WAIT_ASYNC 1
#else
#define HAVE_WAIT_ASYNC 0
#endif
#else
#define HAVE_ASYNC_LDS 0
#define HAVE_WAIT_ASYNC 0
#endif

// The async-to-LDS builtin takes (int4 AS1*, int4 AS3*, imm offset, imm cpol):
// clang diagnostic showed param type 'vector_size(16) int __device__ *'.
typedef __attribute__((__vector_size__(16))) int i4v;
typedef __attribute__((address_space(1))) i4v gas_i4;
typedef __attribute__((address_space(3))) i4v las_i4;
#define TO_GLOBAL(p) ((gas_i4*)(unsigned long long)(p))
#define TO_LDS(p)    ((las_i4*)(unsigned long long)(p))

static __device__ __forceinline__ v16h cat16(v8h lo, v8h hi) {
    return __builtin_shufflevector(lo, hi, 0,1,2,3,4,5,6,7,8,9,10,11,12,13,14,15);
}

// ---------------------------------------------------------------------------
// Kernel 1: seq = x @ W + b  (f16 WMMA, fp32 accum), then RoPE, emit f16 Q/K.
// Block tile: 64(M) x 128(N); wave w -> 16x64 sub-tile (4 accumulators).
// ---------------------------------------------------------------------------
__global__ __launch_bounds__(256) void proj_rope_kernel(
    const float* __restrict__ x, const float* __restrict__ W,
    const float* __restrict__ bias,
    _Float16* __restrict__ q_ws, _Float16* __restrict__ k_ws)
{
    __shared__ union {
        struct { _Float16 A[64][40]; _Float16 Bt[128][40]; } p;  // GEMM tiles
        float seq[64][130];                                      // epilogue tile
    } sh;

    const int t    = threadIdx.x;
    const int lane = t & 31;
    const int wave = t >> 5;
    const int mBase = blockIdx.y * 64;
    const int nBase = blockIdx.x * 128;
    const int mw = (wave >> 1) * 16;     // wave row offset in tile
    const int nw = (wave &  1) * 64;     // wave col offset in tile
    const int g  = lane >> 4;            // half-wave group

    v8f acc[4] = {{}, {}, {}, {}};

    // tile-load work assignment
    const int ar  = t >> 2;              // A row     0..63
    const int ac  = (t & 3) * 8;         // A k-col   0,8,16,24
    const int bp  = t >> 4;              // B k-pair  0..15  (k = 2*bp, 2*bp+1)
    const int bnc = (t & 15) * 8;        // B n-chunk 0..120

    const float* ag = x + (size_t)(mBase + ar) * K_DIM + ac;

    for (int k0 = 0; k0 < K_DIM; k0 += 32) {
        // ---- stage A tile (64x32) as f16, row-major ----
        {
            f4 a0 = *(const f4*)(ag + k0);
            f4 a1 = *(const f4*)(ag + k0 + 4);
            v8h ah;
            ah[0]=(_Float16)a0[0]; ah[1]=(_Float16)a0[1];
            ah[2]=(_Float16)a0[2]; ah[3]=(_Float16)a0[3];
            ah[4]=(_Float16)a1[0]; ah[5]=(_Float16)a1[1];
            ah[6]=(_Float16)a1[2]; ah[7]=(_Float16)a1[3];
            *(v8h*)&sh.p.A[ar][ac] = ah;
        }
        // ---- stage B tile (32x128) transposed -> Bt[n][k]; pack k-pairs ----
        {
            const float* b0 = W + (size_t)(k0 + 2 * bp) * N_DIM + nBase + bnc;
            const float* b1 = b0 + N_DIM;
            f4 r0a = *(const f4*)b0;       f4 r0b = *(const f4*)(b0 + 4);
            f4 r1a = *(const f4*)b1;       f4 r1b = *(const f4*)(b1 + 4);
            #pragma unroll
            for (int j = 0; j < 4; ++j) {
                h2 p0; p0[0] = (_Float16)r0a[j]; p0[1] = (_Float16)r1a[j];
                *(h2*)&sh.p.Bt[bnc + j][2 * bp] = p0;
                h2 p1; p1[0] = (_Float16)r0b[j]; p1[1] = (_Float16)r1b[j];
                *(h2*)&sh.p.Bt[bnc + 4 + j][2 * bp] = p1;
            }
        }
        // prefetch next K-step tiles while this one is consumed
        if (k0 + 32 < K_DIM) {
            __builtin_prefetch(ag + k0 + 32, 0, 1);
            __builtin_prefetch(W + (size_t)(k0 + 32 + 2 * bp) * N_DIM + nBase + bnc, 0, 1);
        }
        __syncthreads();

        // ---- fragments per ISA 16-bit layouts, 4 WMMAs per wave ----
        const int am = mw + (lane & 15);
        v16h afrag = cat16(*(const v8h*)&sh.p.A[am][8 * g],
                           *(const v8h*)&sh.p.A[am][16 + 8 * g]);
        #pragma unroll
        for (int j = 0; j < 4; ++j) {
            const int bn = nw + 16 * j + (lane & 15);
            v16h bf = cat16(*(const v8h*)&sh.p.Bt[bn][8 * g],
                            *(const v8h*)&sh.p.Bt[bn][16 + 8 * g]);
            acc[j] = __builtin_amdgcn_wmma_f32_16x16x32_f16(
                false, afrag, false, bf, (short)0, acc[j], false, false);
        }
        __syncthreads();
    }

    // ---- park fp32 tile in LDS so RoPE pairs (d, d+1) land in one thread ----
    {
        const int row0 = mw + 8 * g;
        const int col  = lane & 15;
        #pragma unroll
        for (int j = 0; j < 4; ++j)
            #pragma unroll
            for (int i = 0; i < 8; ++i)
                sh.seq[row0 + i][nw + 16 * j + col] = acc[j][i];
    }
    __syncthreads();

    // ---- RoPE + scatter to Q/K (f16): 64 rows x 64 pairs = 4096 pairs ----
    const float NEG_LOG1E4_OVER_32 = -0.28782313662425574f; // -ln(10000)/32
    #pragma unroll
    for (int it = 0; it < 16; ++it) {
        const int p    = t + it * 256;
        const int row  = p >> 6;             // 64 pairs per 128-wide row
        const int col  = (p & 63) * 2;
        const int gm   = mBase + row;
        const int bidx = gm >> 10;
        const int pos  = gm & (L_DIM - 1);
        const int gn   = nBase + col;
        const int head = gn >> 7;            // 128 cols per head (q|k)
        const int comp = gn & 127;
        const int d    = comp & (HEAD_SIZE - 1);

        float s0 = sh.seq[row][col]     + bias[gn];
        float s1 = sh.seq[row][col + 1] + bias[gn + 1];

        float ang = (float)pos * __expf((float)(d >> 1) * NEG_LOG1E4_OVER_32);
        float sn, cs;
        __sincosf(ang, &sn, &cs);
        float r0 = s0 * cs - s1 * sn;
        float r1 = s1 * cs + s0 * sn;

        _Float16* dst = ((comp & HEAD_SIZE) ? k_ws : q_ws)
            + (((size_t)(bidx * HEADS + head) * L_DIM + pos) * HEAD_SIZE + d);
        h2 o; o[0] = (_Float16)r0; o[1] = (_Float16)r1;
        *(h2*)dst = o;
    }
}

// ---------------------------------------------------------------------------
// Kernel 2: logits[bh] = (Q Kt - masks) / 8  (f16 WMMA, fp32 accum)
// Grid: (L/128, L/64, B*HEADS). Block tile 64(M) x 128(N); wave -> 16x64.
// Tile staging uses GLOBAL_LOAD_ASYNC_TO_LDS when the toolchain exposes it.
// ---------------------------------------------------------------------------
__global__ __launch_bounds__(256) void qk_logits_kernel(
    const _Float16* __restrict__ q_ws, const _Float16* __restrict__ k_ws,
    const int* __restrict__ mask, float* __restrict__ out)
{
    __shared__ _Float16 Qs[64][72];
    __shared__ _Float16 Ks[128][72];  // n-major x d == Bt layout already

    const int t    = threadIdx.x;
    const int lane = t & 31;
    const int wave = t >> 5;
    const int bh   = blockIdx.z;
    const int b    = bh / HEADS;
    const int mBase = blockIdx.y * 64;
    const int nBase = blockIdx.x * 128;

    // ---- stage Q tile (64x64 halves) and K tile (128x64 halves) ----
    {
        const int r   = t >> 2;          // Q row 0..63
        const int c16 = (t & 3) * 16;    // 16 halves (32 B)
        const _Float16* qg = q_ws + ((size_t)bh * L_DIM + mBase + r) * HEAD_SIZE + c16;
        const int r2  = t >> 1;          // K row 0..127
        const int c32 = (t & 1) * 32;    // 32 halves (64 B)
        const _Float16* kg = k_ws + ((size_t)bh * L_DIM + nBase + r2) * HEAD_SIZE + c32;
#if HAVE_ASYNC_LDS
        __builtin_amdgcn_global_load_async_to_lds_b128(
            TO_GLOBAL(qg),     TO_LDS(&Qs[r][c16]),     0, 0);
        __builtin_amdgcn_global_load_async_to_lds_b128(
            TO_GLOBAL(qg + 8), TO_LDS(&Qs[r][c16 + 8]), 0, 0);
        #pragma unroll
        for (int j = 0; j < 4; ++j)
            __builtin_amdgcn_global_load_async_to_lds_b128(
                TO_GLOBAL(kg + 8 * j), TO_LDS(&Ks[r2][c32 + 8 * j]), 0, 0);
#if HAVE_WAIT_ASYNC
        __builtin_amdgcn_s_wait_asynccnt(0);
#else
        asm volatile("s_wait_asynccnt 0x0" ::: "memory");
#endif
#else
        *(v8h*)&Qs[r][c16]     = *(const v8h*)qg;
        *(v8h*)&Qs[r][c16 + 8] = *(const v8h*)(qg + 8);
        #pragma unroll
        for (int j = 0; j < 4; ++j)
            *(v8h*)&Ks[r2][c32 + 8 * j] = *(const v8h*)(kg + 8 * j);
#endif
    }
    __syncthreads();

    const int mw = (wave >> 1) * 16;
    const int nw = (wave &  1) * 64;
    const int g  = lane >> 4;

    v8f acc[4] = {{}, {}, {}, {}};
    #pragma unroll
    for (int k0 = 0; k0 < HEAD_SIZE; k0 += 32) {
        const int am = mw + (lane & 15);
        v16h afrag = cat16(*(const v8h*)&Qs[am][k0 + 8 * g],
                           *(const v8h*)&Qs[am][k0 + 16 + 8 * g]);
        #pragma unroll
        for (int j = 0; j < 4; ++j) {
            const int bn = nw + 16 * j + (lane & 15);
            v16h bf = cat16(*(const v8h*)&Ks[bn][k0 + 8 * g],
                            *(const v8h*)&Ks[bn][k0 + 16 + 8 * g]);
            acc[j] = __builtin_amdgcn_wmma_f32_16x16x32_f16(
                false, afrag, false, bf, (short)0, acc[j], false, false);
        }
    }

    // ---- epilogue: (raw - pad/causal penalties) * (1/sqrt(64)) ----
    const int row0 = mw + 8 * g;
    const int coln = lane & 15;
    int   ncol[4];
    float colpen[4];
    #pragma unroll
    for (int j = 0; j < 4; ++j) {
        ncol[j]   = nBase + nw + 16 * j + coln;
        colpen[j] = (1.0f - (float)mask[b * L_DIM + ncol[j]]) * MASK_VALF;
    }

    #pragma unroll
    for (int i = 0; i < 8; ++i) {
        const int m = mBase + row0 + i;
        const float rowpen = (1.0f - (float)mask[b * L_DIM + m]) * MASK_VALF;
        const size_t base = ((size_t)bh * L_DIM + m) * (size_t)L_DIM;
        #pragma unroll
        for (int j = 0; j < 4; ++j) {
            float pen = rowpen + colpen[j] + ((ncol[j] < m) ? MASK_VALF : 0.0f);
            out[base + ncol[j]] = (acc[j][i] - pen) * 0.125f;
        }
    }
}

// ---------------------------------------------------------------------------
extern "C" void kernel_launch(void* const* d_in, const int* in_sizes, int n_in,
                              void* d_out, int out_size, void* d_ws, size_t ws_size,
                              hipStream_t stream) {
    (void)in_sizes; (void)n_in; (void)out_size; (void)ws_size;
    const float* x    = (const float*)d_in[0];
    const float* W    = (const float*)d_in[1];
    const float* bias = (const float*)d_in[2];
    const int*   mask = (const int*)d_in[3];
    float* out = (float*)d_out;

    _Float16* q_ws = (_Float16*)d_ws;
    _Float16* k_ws = q_ws + (size_t)B_DIM * HEADS * L_DIM * HEAD_SIZE;

    dim3 g1(N_DIM / 128, M_DIM / 64);                 // (12, 128)
    proj_rope_kernel<<<g1, 256, 0, stream>>>(x, W, bias, q_ws, k_ws);

    dim3 g2(L_DIM / 128, L_DIM / 64, B_DIM * HEADS);  // (8, 16, 96)
    qk_logits_kernel<<<g2, 256, 0, stream>>>(q_ws, k_ws, mask, out);
}
